// BarkSelfAttention_23742579212384
// MI455X (gfx1250) — compile-verified, matrix operand order checked by reading, not run on hardware
//
#include <hip/hip_runtime.h>
#include <hip/hip_bf16.h>
#include <cstdint>

typedef __attribute__((ext_vector_type(16))) __bf16 v16bf;
typedef __attribute__((ext_vector_type(8)))  float  v8f;

union Frag {
  v16bf v;
  uint4 q[2];
  unsigned short h[16];
};

union U4H8 {
  uint4 q;
  unsigned short h[8];
};

__device__ __forceinline__ unsigned short f2bf(float f) {
  unsigned u = __float_as_uint(f);
  u += 0x7FFFu + ((u >> 16) & 1u);   // round-to-nearest-even
  return (unsigned short)(u >> 16);
}

__device__ __forceinline__ v8f wmma_bf16(v16bf a, v16bf b, v8f c) {
  return __builtin_amdgcn_wmma_f32_16x16x32_bf16(false, a, false, b, (short)0, c,
                                                 false, false);
}

// Async DMA: copy 16 bytes global -> LDS per lane (ASYNCcnt-tracked).
// VDST = per-lane LDS byte address; flat shared-aperture addresses carry the
// wave-relative LDS offset in bits [31:0] (ISA 10.2 aperture mapping).
__device__ __forceinline__ void async_copy16(void* ldsDst, const void* src) {
  unsigned off = (unsigned)(uintptr_t)ldsDst;
  asm volatile("global_load_async_to_lds_b128 %0, %1, off"
               :: "v"(off), "v"(src)
               : "memory");
}
#define WAIT_ASYNC(n) asm volatile("s_wait_asynccnt " #n ::: "memory")

// ---------------------------------------------------------------------------
// Pre-pass 1: X f32 -> bf16 row-major.
// ---------------------------------------------------------------------------
__global__ __launch_bounds__(256) void convert_x(const float* __restrict__ X,
                                                 unsigned short* __restrict__ Xb)
{
  size_t idx = ((size_t)blockIdx.x * 256 + threadIdx.x) * 8;
  float4 a = *(const float4*)(X + idx);
  float4 b = *(const float4*)(X + idx + 4);
  U4H8 o;
  o.h[0] = f2bf(a.x); o.h[1] = f2bf(a.y); o.h[2] = f2bf(a.z); o.h[3] = f2bf(a.w);
  o.h[4] = f2bf(b.x); o.h[5] = f2bf(b.y); o.h[6] = f2bf(b.z); o.h[7] = f2bf(b.w);
  *(uint4*)(Xb + idx) = o.q;
}

// ---------------------------------------------------------------------------
// Pre-pass 2: W[K=1024][N] f32 -> Wt[N][1024] bf16 (transpose via LDS tile).
// ---------------------------------------------------------------------------
__global__ __launch_bounds__(256) void wtrans(const float* __restrict__ W,
                                              unsigned short* __restrict__ Wt,
                                              int N)
{
  __shared__ unsigned short lt[32][33];
  const int tid = threadIdx.x;
  const int n0 = blockIdx.x * 32, k0 = blockIdx.y * 32;
  #pragma unroll
  for (int c = 0; c < 4; ++c) {
    int flat = c * 256 + tid;
    int kr = flat >> 5, nc = flat & 31;
    lt[nc][kr] = f2bf(W[(size_t)(k0 + kr) * N + n0 + nc]);
  }
  __syncthreads();
  #pragma unroll
  for (int c = 0; c < 4; ++c) {
    int flat = c * 256 + tid;
    int nr = flat >> 5, kc = flat & 31;
    Wt[(size_t)(n0 + nr) * 1024 + k0 + kc] = lt[nr][kc];
  }
}

// ---------------------------------------------------------------------------
// Kernel 1: QKV projection, all-bf16 operands, async double-buffered staging.
// C[8192,3072] = Xb @ Wt^T + b. q scaled by 1/8; v written transposed.
// Block tile 64x64, 4 waves, each wave a 32x32 quadrant (2x2 WMMA).
// ---------------------------------------------------------------------------
__global__ __launch_bounds__(128) void qkv_gemm(
    const unsigned short* __restrict__ Xb,
    const unsigned short* __restrict__ Wt,   // [3072][1024] bf16
    const float* __restrict__ bias,
    unsigned short* __restrict__ qb, unsigned short* __restrict__ kb,
    unsigned short* __restrict__ vt)
{
  __shared__ unsigned short ldsA[2][64][32];   // [m][k]
  __shared__ unsigned short ldsBt[2][64][32];  // [n][k]
  const int tid  = threadIdx.x;
  const int lane = tid & 31;
  const int wave = tid >> 5;
  const int wr = wave >> 1, wc = wave & 1;
  const int rowBase = blockIdx.y * 64;
  const int colBase = blockIdx.x * 64;
  const int hh = lane >> 4, h8 = hh * 8, l15 = lane & 15;

  v8f acc[2][2] = {};

  auto stage = [&](int t) {
    int buf = t & 1;
    int k0 = t * 32;
    #pragma unroll
    for (int c = 0; c < 2; ++c) {
      int flat = (c * 128 + tid) * 8;
      int ar = flat >> 5, ac = flat & 31;
      async_copy16(&ldsA[buf][ar][ac],
                   Xb + (size_t)(rowBase + ar) * 1024 + k0 + ac);
    }
    #pragma unroll
    for (int c = 0; c < 2; ++c) {
      int flat = (c * 128 + tid) * 8;
      int nr = flat >> 5, kc = flat & 31;
      async_copy16(&ldsBt[buf][nr][kc],
                   Wt + (size_t)(colBase + nr) * 1024 + k0 + kc);
    }
  };

  stage(0);
  for (int t = 0; t < 32; ++t) {
    const int buf = t & 1;
    if (t < 31) { stage(t + 1); WAIT_ASYNC(4); }
    else        { WAIT_ASYNC(0); }
    __syncthreads();

    Frag a[2], b[2];
    #pragma unroll
    for (int mi = 0; mi < 2; ++mi) {
      int r = wr * 32 + mi * 16 + l15;
      a[mi].q[0] = *(const uint4*)&ldsA[buf][r][h8];
      a[mi].q[1] = *(const uint4*)&ldsA[buf][r][16 + h8];
    }
    #pragma unroll
    for (int ni = 0; ni < 2; ++ni) {
      int n = wc * 32 + ni * 16 + l15;
      b[ni].q[0] = *(const uint4*)&ldsBt[buf][n][h8];
      b[ni].q[1] = *(const uint4*)&ldsBt[buf][n][16 + h8];
    }
    #pragma unroll
    for (int mi = 0; mi < 2; ++mi)
      #pragma unroll
      for (int ni = 0; ni < 2; ++ni)
        acc[mi][ni] = wmma_bf16(a[mi].v, b[ni].v, acc[mi][ni]);
    __syncthreads();
  }

  // epilogue: bias, q-scale, scatter. q/k row-major [B,H,S,64]; v transposed
  // [B,H,64,S] so the attention V staging is a pure async copy.
  #pragma unroll
  for (int mi = 0; mi < 2; ++mi)
    #pragma unroll
    for (int ni = 0; ni < 2; ++ni) {
      int n = colBase + wc * 32 + ni * 16 + l15;
      float bv = bias[n];
      int seg = n >> 10;
      int nl = n & 1023;
      int head = nl >> 6, dv = nl & 63;
      float scl = (seg == 0) ? 0.125f : 1.0f;
      #pragma unroll
      for (int i = 0; i < 8; ++i) {
        int m = rowBase + wr * 32 + mi * 16 + i + 8 * hh;
        int bidx = m >> 10, s = m & 1023;
        unsigned short bf = f2bf((acc[mi][ni][i] + bv) * scl);
        size_t bh = (size_t)(bidx * 16 + head);
        if (seg == 2) {
          vt[(bh * 64 + dv) * 1024 + s] = bf;
        } else {
          unsigned short* dst = (seg == 0) ? qb : kb;
          dst[(bh * 1024 + s) * 64 + dv] = bf;
        }
      }
    }
}

// ---------------------------------------------------------------------------
// Kernel 2: flash attention. One wave per (batch, head, 16-query tile).
// V^T blocks double-buffered into LDS via async DMA; online softmax.
// ---------------------------------------------------------------------------
__global__ __launch_bounds__(32) void attn_kernel(
    const unsigned short* __restrict__ qb,
    const unsigned short* __restrict__ kbuf,
    const unsigned short* __restrict__ vt,   // [B*H][64][1024]
    unsigned short* __restrict__ ob)
{
  __shared__ unsigned short ldsVt[2][64][32];  // [dv][key]
  __shared__ unsigned short ldsP[16][32];      // [m][key]
  const int lane = threadIdx.x;
  const int l15 = lane & 15, hh = lane >> 4, h8 = hh * 8;
  const int qt = blockIdx.x, head = blockIdx.y, bat = blockIdx.z;
  const int bh = bat * 16 + head;
  const int q0 = qt * 16;
  const size_t headBase = (size_t)bh * 1024 * 64;
  const size_t vtBase   = (size_t)bh * 64 * 1024;

  // Q tile 16x64 in A-layout (q pre-scaled by 1/8)
  Frag aQ[2];
  {
    size_t rb = headBase + (size_t)(q0 + l15) * 64;
    #pragma unroll
    for (int c = 0; c < 2; ++c) {
      aQ[c].q[0] = *(const uint4*)(qb + rb + c * 32 + h8);
      aQ[c].q[1] = *(const uint4*)(qb + rb + c * 32 + 16 + h8);
    }
  }

  float mrow[8], lrow[8];
  v8f accO[4] = {};
  #pragma unroll
  for (int i = 0; i < 8; ++i) { mrow[i] = -1e30f; lrow[i] = 0.0f; }

  auto stageV = [&](int t) {
    int buf = t & 1;
    int key0 = t * 32;
    #pragma unroll
    for (int c = 0; c < 8; ++c) {
      int flat = (c * 32 + lane) * 8;
      int dvr = flat >> 5, kc = flat & 31;
      async_copy16(&ldsVt[buf][dvr][kc],
                   vt + vtBase + (size_t)dvr * 1024 + key0 + kc);
    }
  };

  const int nkb = (q0 + 16 + 31) >> 5;   // causal: keys <= q0+15
  stageV(0);
  for (int t = 0; t < nkb; ++t) {
    const int buf = t & 1;
    const int key0 = t * 32;
    if (t + 1 < nkb) { stageV(t + 1); WAIT_ASYNC(8); }
    else             { WAIT_ASYNC(0); }
    __syncthreads();

    // scores S = Q @ K^T, two 16x16 column tiles (K read straight from HBM)
    v8f sj[2];
    #pragma unroll
    for (int j = 0; j < 2; ++j) {
      int key = key0 + j * 16 + l15;
      size_t rb = headBase + (size_t)key * 64;
      Frag bK[2];
      #pragma unroll
      for (int c = 0; c < 2; ++c) {
        bK[c].q[0] = *(const uint4*)(kbuf + rb + c * 32 + h8);
        bK[c].q[1] = *(const uint4*)(kbuf + rb + c * 32 + 16 + h8);
      }
      v8f s = {};
      s = wmma_bf16(aQ[0].v, bK[0].v, s);
      s = wmma_bf16(aQ[1].v, bK[1].v, s);
      #pragma unroll
      for (int i = 0; i < 8; ++i) {
        int mr = q0 + i + 8 * hh;
        if (key > mr) s[i] = -1e30f;   // causal mask
      }
      sj[j] = s;
    }

    // online softmax (row reductions across 16-lane halves)
    #pragma unroll
    for (int i = 0; i < 8; ++i) {
      float v = fmaxf(sj[0][i], sj[1][i]);
      v = fmaxf(v, __shfl_xor(v, 1, 32));
      v = fmaxf(v, __shfl_xor(v, 2, 32));
      v = fmaxf(v, __shfl_xor(v, 4, 32));
      v = fmaxf(v, __shfl_xor(v, 8, 32));
      float mn = fmaxf(mrow[i], v);
      float alpha = __expf(mrow[i] - mn);
      mrow[i] = mn;
      float p0 = __expf(sj[0][i] - mn);
      float p1 = __expf(sj[1][i] - mn);
      float rs = p0 + p1;
      rs += __shfl_xor(rs, 1, 32);
      rs += __shfl_xor(rs, 2, 32);
      rs += __shfl_xor(rs, 4, 32);
      rs += __shfl_xor(rs, 8, 32);
      lrow[i] = lrow[i] * alpha + rs;
      #pragma unroll
      for (int jn = 0; jn < 4; ++jn) accO[jn][i] *= alpha;
      int ml = i + 8 * hh;
      ldsP[ml][l15]      = f2bf(p0);   // C-layout -> A-layout via LDS
      ldsP[ml][16 + l15] = f2bf(p1);
    }
    __syncthreads();

    // O += P(16x32) @ V(32x64)
    Frag aP;
    aP.q[0] = *(const uint4*)&ldsP[l15][h8];
    aP.q[1] = *(const uint4*)&ldsP[l15][16 + h8];
    #pragma unroll
    for (int jn = 0; jn < 4; ++jn) {
      Frag bV;
      int dv = jn * 16 + l15;
      bV.q[0] = *(const uint4*)&ldsVt[buf][dv][h8];
      bV.q[1] = *(const uint4*)&ldsVt[buf][dv][16 + h8];
      accO[jn] = wmma_bf16(aP.v, bV.v, accO[jn]);
    }
    __syncthreads();
  }

  // normalize, store to [B,S,H*64] bf16
  #pragma unroll
  for (int jn = 0; jn < 4; ++jn)
    #pragma unroll
    for (int i = 0; i < 8; ++i) {
      int mr = q0 + i + 8 * hh;
      int dv = jn * 16 + l15;
      float o = accO[jn][i] / lrow[i];
      ob[((size_t)bat * 1024 + mr) * 1024 + head * 64 + dv] = f2bf(o);
    }
}

// ---------------------------------------------------------------------------
// Kernel 3: output projection. out[8192,1024] = O_bf16 @ Wt_out^T + b_out.
// ---------------------------------------------------------------------------
__global__ __launch_bounds__(128) void out_gemm(
    const unsigned short* __restrict__ A,
    const unsigned short* __restrict__ Wt,   // [1024][1024] bf16
    const float* __restrict__ bias, float* __restrict__ out)
{
  __shared__ unsigned short ldsA[2][64][32];
  __shared__ unsigned short ldsBt[2][64][32];
  const int tid  = threadIdx.x;
  const int lane = tid & 31;
  const int wave = tid >> 5;
  const int wr = wave >> 1, wc = wave & 1;
  const int rowBase = blockIdx.y * 64;
  const int colBase = blockIdx.x * 64;
  const int hh = lane >> 4, h8 = hh * 8, l15 = lane & 15;

  v8f acc[2][2] = {};

  auto stage = [&](int t) {
    int buf = t & 1;
    int k0 = t * 32;
    #pragma unroll
    for (int c = 0; c < 2; ++c) {
      int flat = (c * 128 + tid) * 8;
      int ar = flat >> 5, ac = flat & 31;
      async_copy16(&ldsA[buf][ar][ac],
                   A + (size_t)(rowBase + ar) * 1024 + k0 + ac);
    }
    #pragma unroll
    for (int c = 0; c < 2; ++c) {
      int flat = (c * 128 + tid) * 8;
      int nr = flat >> 5, kc = flat & 31;
      async_copy16(&ldsBt[buf][nr][kc],
                   Wt + (size_t)(colBase + nr) * 1024 + k0 + kc);
    }
  };

  stage(0);
  for (int t = 0; t < 32; ++t) {
    const int buf = t & 1;
    if (t < 31) { stage(t + 1); WAIT_ASYNC(4); }
    else        { WAIT_ASYNC(0); }
    __syncthreads();

    Frag a[2], b[2];
    #pragma unroll
    for (int mi = 0; mi < 2; ++mi) {
      int r = wr * 32 + mi * 16 + l15;
      a[mi].q[0] = *(const uint4*)&ldsA[buf][r][h8];
      a[mi].q[1] = *(const uint4*)&ldsA[buf][r][16 + h8];
    }
    #pragma unroll
    for (int ni = 0; ni < 2; ++ni) {
      int n = wc * 32 + ni * 16 + l15;
      b[ni].q[0] = *(const uint4*)&ldsBt[buf][n][h8];
      b[ni].q[1] = *(const uint4*)&ldsBt[buf][n][16 + h8];
    }
    #pragma unroll
    for (int mi = 0; mi < 2; ++mi)
      #pragma unroll
      for (int ni = 0; ni < 2; ++ni)
        acc[mi][ni] = wmma_bf16(a[mi].v, b[ni].v, acc[mi][ni]);
    __syncthreads();
  }

  #pragma unroll
  for (int mi = 0; mi < 2; ++mi)
    #pragma unroll
    for (int ni = 0; ni < 2; ++ni) {
      int n = colBase + wc * 32 + ni * 16 + l15;
      float bv = bias[n];
      #pragma unroll
      for (int i = 0; i < 8; ++i) {
        int m = rowBase + wr * 32 + mi * 16 + i + 8 * hh;
        out[(size_t)m * 1024 + n] = acc[mi][ni][i] + bv;
      }
    }
}

// ---------------------------------------------------------------------------
extern "C" void kernel_launch(void* const* d_in, const int* in_sizes, int n_in,
                              void* d_out, int out_size, void* d_ws, size_t ws_size,
                              hipStream_t stream) {
  const float* X      = (const float*)d_in[0];   // [8,1024,1024]
  const float* W_attn = (const float*)d_in[1];   // [1024,3072]
  const float* b_attn = (const float*)d_in[2];   // [3072]
  const float* W_out  = (const float*)d_in[3];   // [1024,1024]
  const float* b_out  = (const float*)d_in[4];   // [1024]
  float* out = (float*)d_out;

  const size_t E = (size_t)8 * 16 * 1024 * 64;   // 8,388,608
  unsigned short* qb  = (unsigned short*)d_ws;   // [B,H,S,64]
  unsigned short* kb  = qb + E;                  // [B,H,S,64]
  unsigned short* vt  = kb + E;                  // [B,H,64,S]
  unsigned short* ob  = vt + E;                  // [8192,1024]
  unsigned short* xb  = ob + E;                  // [8192,1024]
  unsigned short* wta = xb + E;                  // [3072,1024]
  unsigned short* wto = wta + (size_t)3072 * 1024; // [1024,1024]

  convert_x<<<dim3(4096), dim3(256), 0, stream>>>(X, xb);
  wtrans<<<dim3(96, 32), dim3(256), 0, stream>>>(W_attn, wta, 3072);
  wtrans<<<dim3(32, 32), dim3(256), 0, stream>>>(W_out, wto, 1024);
  qkv_gemm<<<dim3(48, 128), dim3(128), 0, stream>>>(xb, wta, b_attn, qb, kb, vt);
  attn_kernel<<<dim3(64, 16, 8), dim3(32), 0, stream>>>(qb, kb, vt, ob);
  out_gemm<<<dim3(16, 128), dim3(128), 0, stream>>>(ob, wto, b_out, out);
}
